// PiTBlock_35691178230025
// MI455X (gfx1250) — matrix operand order repaired and checked by reading, not compiled, not run on hardware
//
#include <hip/hip_runtime.h>
#include <hip/hip_bf16.h>

// ---------------------------------------------------------------------------
// Problem constants (from reference): B=4, N=1024, H=1024, A=1024, NH=16,
// HD=64, FFN=2730 (padded to 2752 so K is a multiple of 32 for WMMA).
// ---------------------------------------------------------------------------
#define CB   4
#define CN   1024
#define CH   1024
#define CA   1024
#define CNH  16
#define CHD  64
#define CM   (CB * CN)        // 4096 token rows
#define CF   2730
#define CFP  2752
#define C2F  (2 * CF)         // 5460
#define SIXH (6 * CH)
#define CEPS 1e-6f

typedef __bf16 bf16_t;
typedef __attribute__((ext_vector_type(16))) __bf16 v16bf;
typedef __attribute__((ext_vector_type(8)))  __bf16 v8bf;
typedef __attribute__((ext_vector_type(8)))  float  v8f;
typedef int v4i_vs __attribute__((vector_size(16)));   // matches builtin's V4i

// --- CDNA5 async global->LDS path (guarded; fallback is sw-pipelined) ------
#if __has_builtin(__builtin_amdgcn_global_load_async_to_lds_b128)
#define HAVE_ASYNC_LDS 1
#define ASYNC_CP16(dst_lds, src_glb)                                           \
    __builtin_amdgcn_global_load_async_to_lds_b128(                            \
        (__attribute__((address_space(1))) v4i_vs*)(uintptr_t)(src_glb),       \
        (__attribute__((address_space(3))) v4i_vs*)(uint32_t)(uintptr_t)(dst_lds), \
        0, 0)
#else
#define HAVE_ASYNC_LDS 0
#endif

static __device__ __forceinline__ void wait_async() {
#if HAVE_ASYNC_LDS
#if __has_builtin(__builtin_amdgcn_s_wait_asynccnt)
    __builtin_amdgcn_s_wait_asynccnt(0);
#else
    asm volatile("s_wait_asynccnt 0x0" ::: "memory");
#endif
#endif
}

static __device__ __forceinline__ v8f vzero8f() {
    v8f z;
#pragma unroll
    for (int i = 0; i < 8; ++i) z[i] = 0.0f;
    return z;
}
static __device__ __forceinline__ v16bf cat16(v8bf lo, v8bf hi) {
    v16bf r;
#pragma unroll
    for (int i = 0; i < 8; ++i) { r[i] = lo[i]; r[i + 8] = hi[i]; }
    return r;
}

// ---------------------------------------------------------------------------
// Small elementwise / GEMV kernels (bandwidth-trivial, stay fp32)
// ---------------------------------------------------------------------------
__global__ void k_silu(const float* __restrict__ c, float* __restrict__ sc, int n) {
    int i = blockIdx.x * blockDim.x + threadIdx.x;
    if (i < n) { float v = c[i]; sc[i] = v / (1.0f + __expf(-v)); }
}

// ada[b, j] = silu(c[b]) . ada_w[j] + ada_b[j]   (B*6H = 24576 dots of 1024)
__global__ void k_ada(const float* __restrict__ sc, const float* __restrict__ w,
                      const float* __restrict__ bias, float* __restrict__ out) {
    int o = blockIdx.x * blockDim.x + threadIdx.x;   // [0, B*6H)
    int b = o / SIXH, j = o % SIXH;
    const float* cr = sc + (size_t)b * CH;
    const float* wr = w + (size_t)j * CH;
    float s = bias[j];
#pragma unroll 4
    for (int i = 0; i < CH; ++i) s += cr[i] * wr[i];
    out[o] = s;
}

__global__ void k_cvt(const float* __restrict__ s, bf16_t* __restrict__ d, int n) {
    int i = blockIdx.x * blockDim.x + threadIdx.x;
    if (i < n) d[i] = (bf16_t)s[i];
}

// Convert [rows, K] fp32 -> [rows, Kp] bf16 with zero K-padding (w3 weight)
__global__ void k_cvt_pad(const float* __restrict__ s, bf16_t* __restrict__ d,
                          int rows, int K, int Kp) {
    int i = blockIdx.x * blockDim.x + threadIdx.x;
    if (i >= rows * Kp) return;
    int r = i / Kp, j = i % Kp;
    d[i] = (j < K) ? (bf16_t)s[(size_t)r * K + j] : (bf16_t)0.0f;
}

// h[m,:] = (rms(x[m,:]) * normw) * (1 + ada[b, scaleCol + :]) + ada[b, shiftCol + :]
__global__ __launch_bounds__(256)
void k_modulate(const float* __restrict__ X, const float* __restrict__ normw,
                const float* __restrict__ ada, int shiftCol, int scaleCol,
                bf16_t* __restrict__ Hout) {
    int m = blockIdx.x;
    int tid = threadIdx.x, lane = tid & 31, wave = tid >> 5;
    const float* xr = X + (size_t)m * CH;
    const float* ar = ada + (size_t)(m / CN) * SIXH;
    float s = 0.0f;
    for (int j = tid; j < CH; j += 256) { float v = xr[j]; s += v * v; }
#pragma unroll
    for (int off = 16; off >= 1; off >>= 1) s += __shfl_xor(s, off, 32);
    __shared__ float red[8];
    if (lane == 0) red[wave] = s;
    __syncthreads();
    float tot = 0.0f;
#pragma unroll
    for (int i = 0; i < 8; ++i) tot += red[i];
    float inv = rsqrtf(tot / (float)CH + CEPS);
    for (int j = tid; j < CH; j += 256) {
        float hn = xr[j] * inv * normw[j];
        float o = hn * (1.0f + ar[scaleCol + j]) + ar[shiftCol + j];
        Hout[(size_t)m * CH + j] = (bf16_t)o;
    }
}

// ---------------------------------------------------------------------------
// Generic bf16 WMMA GEMM:  out[m,n] = A[m,:K] . W[n,:K] + bias[n]  (+epilogue)
//   A: [M,K] bf16 row-major, W: [Nout,K] bf16 row-major (i.e. B^T)
//   Workgroup tile 128x128, BK=32, 8 waves (2x4), wave tile 64x32 (8 C-tiles)
//   Double-buffered LDS; next tile staged via GLOBAL_LOAD_ASYNC_TO_LDS_B128
//   (ASYNCcnt) when available, else sw-pipelined load->wmma->ds_store.
//   mode 0: store bf16; mode 1: store f32; mode 2: f32 = resid + gate*acc
// ---------------------------------------------------------------------------
#define GBM 128
#define GBN 128
#define GBK 32
#define LDP (GBK + 8)   // LDS row stride (bf16 elems), keeps 16B alignment

__global__ __launch_bounds__(256)
void k_gemm(const bf16_t* __restrict__ A, const bf16_t* __restrict__ W,
            const float* __restrict__ bias, int M, int Nout, int K,
            float* __restrict__ outF, bf16_t* __restrict__ outB,
            const float* __restrict__ resid, const float* __restrict__ gate,
            int gStride, int mode) {
    __shared__ bf16_t Asb[2][GBM][LDP];
    __shared__ bf16_t Wsb[2][GBN][LDP];

    int tid = threadIdx.x;
    int lane = tid & 31;
    int wave = tid >> 5;
    int wm = wave >> 2;          // 0..1 -> 64-row slice
    int wn = wave & 3;           // 0..3 -> 32-col slice
    int l16 = lane & 15;
    int half = (lane >> 4) & 1;  // 0: lanes 0-15, 1: lanes 16-31
    int m0 = blockIdx.y * GBM;
    int n0 = blockIdx.x * GBN;

    v8f acc[4][2];
#pragma unroll
    for (int mi = 0; mi < 4; ++mi)
#pragma unroll
        for (int ni = 0; ni < 2; ++ni) acc[mi][ni] = vzero8f();

    // --- prologue: stage tile 0 into buffer 0 (synchronous) ---
#pragma unroll
    for (int s = 0; s < 2; ++s) {
        int slot = tid + s * 256;
        int r = slot >> 2;
        int kc = (slot & 3) * 8;
        v8bf av = *(const v8bf*)(A + (size_t)(m0 + r) * K + kc);
        *(v8bf*)(&Asb[0][r][kc]) = av;
        int nr = n0 + r; if (nr >= Nout) nr = Nout - 1;   // clamp ragged N
        v8bf wv = *(const v8bf*)(W + (size_t)nr * K + kc);
        *(v8bf*)(&Wsb[0][r][kc]) = wv;
    }
    __syncthreads();

    int nsteps = K / GBK;
    for (int step = 0; step < nsteps; ++step) {
        int cur = step & 1, nxt = cur ^ 1;
        bool hasNext = (step + 1 < nsteps);
        int kn = (step + 1) * GBK;

#if HAVE_ASYNC_LDS
        // --- issue async copies of tile step+1 into the other buffer ---
        if (hasNext) {
#pragma unroll
            for (int s = 0; s < 2; ++s) {
                int slot = tid + s * 256;
                int r = slot >> 2;
                int kc = (slot & 3) * 8;
                ASYNC_CP16(&Asb[nxt][r][kc], A + (size_t)(m0 + r) * K + kn + kc);
                int nr = n0 + r; if (nr >= Nout) nr = Nout - 1;
                ASYNC_CP16(&Wsb[nxt][r][kc], W + (size_t)nr * K + kn + kc);
            }
        }
#else
        // --- sw pipeline: issue global loads now, publish to LDS after wmma ---
        v8bf ta[2], tw[2];
        if (hasNext) {
#pragma unroll
            for (int s = 0; s < 2; ++s) {
                int slot = tid + s * 256;
                int r = slot >> 2;
                int kc = (slot & 3) * 8;
                ta[s] = *(const v8bf*)(A + (size_t)(m0 + r) * K + kn + kc);
                int nr = n0 + r; if (nr >= Nout) nr = Nout - 1;
                tw[s] = *(const v8bf*)(W + (size_t)nr * K + kn + kc);
            }
            if (kn + GBK < K)
                __builtin_prefetch(A + (size_t)(m0 + (tid >> 2)) * K + kn + GBK, 0, 1);
        }
#endif

        // --- B fragments (weight rows = output cols) ---
        v16bf bfrag[2];
#pragma unroll
        for (int ni = 0; ni < 2; ++ni) {
            int wr = wn * 32 + ni * 16 + l16;
            int kb = half * 8;
            v8bf lo = *(const v8bf*)(&Wsb[cur][wr][kb]);
            v8bf hi = *(const v8bf*)(&Wsb[cur][wr][kb + 16]);
            bfrag[ni] = cat16(lo, hi);
        }
        // --- A fragments + 8 WMMAs ---
#pragma unroll
        for (int mi = 0; mi < 4; ++mi) {
            int ar = wm * 64 + mi * 16 + l16;
            int kb = half * 8;
            v8bf lo = *(const v8bf*)(&Asb[cur][ar][kb]);
            v8bf hi = *(const v8bf*)(&Asb[cur][ar][kb + 16]);
            v16bf afrag = cat16(lo, hi);
#pragma unroll
            for (int ni = 0; ni < 2; ++ni)
                acc[mi][ni] = __builtin_amdgcn_wmma_f32_16x16x32_bf16(
                    false, afrag, false, bfrag[ni], (short)0, acc[mi][ni],
                    false, false);
        }

#if HAVE_ASYNC_LDS
        wait_async();            // own async copies landed in LDS
#else
        if (hasNext) {
#pragma unroll
            for (int s = 0; s < 2; ++s) {
                int slot = tid + s * 256;
                int r = slot >> 2;
                int kc = (slot & 3) * 8;
                *(v8bf*)(&Asb[nxt][r][kc]) = ta[s];
                *(v8bf*)(&Wsb[nxt][r][kc]) = tw[s];
            }
        }
#endif
        __syncthreads();         // all waves done: tile published, cur consumed
    }

    // --- epilogue ---
#pragma unroll
    for (int mi = 0; mi < 4; ++mi) {
#pragma unroll
        for (int ni = 0; ni < 2; ++ni) {
            int n = n0 + wn * 32 + ni * 16 + l16;
            if (n >= Nout) continue;
            float bz = bias ? bias[n] : 0.0f;
            int mbase = m0 + wm * 64 + mi * 16 + half * 8;
#pragma unroll
            for (int v = 0; v < 8; ++v) {
                int m = mbase + v;
                float r = acc[mi][ni][v] + bz;
                size_t idx = (size_t)m * Nout + n;
                if (mode == 0) {
                    outB[idx] = (bf16_t)r;
                } else if (mode == 1) {
                    outF[idx] = r;
                } else {
                    int b = m / CN;
                    float g = gate[(size_t)b * gStride + n];
                    outF[idx] = resid[idx] + g * r;
                }
            }
        }
    }
}

// ---------------------------------------------------------------------------
// q/k RMS head-norm + 1/sqrt(HD) folding; re-layout to [B*NH, N, HD] bf16
// qkv rows: [3][NH][HD] per token (from qkv GEMM, Nout = 3A)
// ---------------------------------------------------------------------------
__global__ __launch_bounds__(32)
void k_qkvprep(const float* __restrict__ qkv, const float* __restrict__ qn,
               const float* __restrict__ kn, bf16_t* __restrict__ Qo,
               bf16_t* __restrict__ Ko, bf16_t* __restrict__ Vo, float scale) {
    int idx = blockIdx.x;            // [0, M*NH)
    int m = idx / CNH, h = idx % CNH;
    int b = m / CN, n = m % CN;
    int lane = threadIdx.x;
    const float* base = qkv + (size_t)m * (3 * CA) + h * CHD;
    float q0 = base[lane],            q1 = base[lane + 32];
    float k0 = base[CA + lane],       k1 = base[CA + lane + 32];
    float v0 = base[2 * CA + lane],   v1 = base[2 * CA + lane + 32];
    float sq = q0 * q0 + q1 * q1;
    float sk = k0 * k0 + k1 * k1;
#pragma unroll
    for (int off = 16; off >= 1; off >>= 1) {
        sq += __shfl_xor(sq, off, 32);
        sk += __shfl_xor(sk, off, 32);
    }
    float rq = rsqrtf(sq / (float)CHD + CEPS) * scale;
    float rk = rsqrtf(sk / (float)CHD + CEPS);
    size_t ob = ((size_t)(b * CNH + h) * CN + n) * CHD;
    Qo[ob + lane]      = (bf16_t)(q0 * rq * qn[lane]);
    Qo[ob + lane + 32] = (bf16_t)(q1 * rq * qn[lane + 32]);
    Ko[ob + lane]      = (bf16_t)(k0 * rk * kn[lane]);
    Ko[ob + lane + 32] = (bf16_t)(k1 * rk * kn[lane + 32]);
    Vo[ob + lane]      = (bf16_t)v0;
    Vo[ob + lane + 32] = (bf16_t)v1;
}

// ---------------------------------------------------------------------------
// Flash attention, bf16 WMMA.  Grid: (B*NH, N/128); 256 threads = 8 waves.
// Each wave owns a 16-row Q tile; key blocks of 32 staged in LDS (V transposed).
// S = Q.K^T (4 wmma), online softmax, P.V (4 wmma). Output -> [M, A] bf16.
// ---------------------------------------------------------------------------
__global__ __launch_bounds__(256)
void k_attn(const bf16_t* __restrict__ Q, const bf16_t* __restrict__ Kk,
            const bf16_t* __restrict__ V, bf16_t* __restrict__ O) {
    __shared__ bf16_t Kb[32][CHD + 8];     // key block, row-major
    __shared__ bf16_t Vt[CHD][32 + 8];     // value block, transposed
    __shared__ bf16_t Pb[8][16][32 + 8];   // per-wave P scratch

    int bh = blockIdx.x;                   // [0, B*NH)
    int qt = blockIdx.y;                   // [0, N/128)
    int b = bh / CNH, h = bh % CNH;
    int tid = threadIdx.x, lane = tid & 31, wave = tid >> 5;
    int l16 = lane & 15;
    int half = (lane >> 4) & 1;
    int qr0 = qt * 128 + wave * 16;
    const size_t headOff = (size_t)bh * CN * CHD;

    // Q fragments for this wave's 16 rows: 2 chunks of K=32 over HD=64
    v16bf qfrag[2];
#pragma unroll
    for (int c = 0; c < 2; ++c) {
        const bf16_t* qp = Q + headOff + (size_t)(qr0 + l16) * CHD + c * 32;
        int kb = half * 8;
        v8bf lo = *(const v8bf*)(qp + kb);
        v8bf hi = *(const v8bf*)(qp + kb + 16);
        qfrag[c] = cat16(lo, hi);
    }

    v8f occ[4];
#pragma unroll
    for (int c = 0; c < 4; ++c) occ[c] = vzero8f();
    float rmax[8], rsum[8];
#pragma unroll
    for (int v = 0; v < 8; ++v) { rmax[v] = -1e30f; rsum[v] = 0.0f; }

    for (int j = 0; j < CN / 32; ++j) {
        // --- stage K block (row-major) and V block (transposed) ---
        {
            int row = tid >> 3;
            int c8 = (tid & 7) * 8;
            v8bf kv = *(const v8bf*)(Kk + headOff + (size_t)(j * 32 + row) * CHD + c8);
            *(v8bf*)(&Kb[row][c8]) = kv;
            v8bf vv = *(const v8bf*)(V + headOff + (size_t)(j * 32 + row) * CHD + c8);
#pragma unroll
            for (int i = 0; i < 8; ++i) Vt[c8 + i][row] = vv[i];
        }
        __syncthreads();

        // --- S tiles: 16 q-rows x 32 keys = two 16x16 f32 tiles ---
        v8f s[2];
#pragma unroll
        for (int t = 0; t < 2; ++t) {
            s[t] = vzero8f();
#pragma unroll
            for (int c = 0; c < 2; ++c) {
                int kr = t * 16 + l16;            // key index in block
                int cb = c * 32 + half * 8;       // HD offset
                v8bf lo = *(const v8bf*)(&Kb[kr][cb]);
                v8bf hi = *(const v8bf*)(&Kb[kr][cb + 16]);
                s[t] = __builtin_amdgcn_wmma_f32_16x16x32_bf16(
                    false, qfrag[c], false, cat16(lo, hi), (short)0, s[t],
                    false, false);
            }
        }

        // --- online softmax (row stats replicated across each 16-lane group) ---
#pragma unroll
        for (int v = 0; v < 8; ++v) {
            float ml = fmaxf(s[0][v], s[1][v]);
#pragma unroll
            for (int off = 1; off < 16; off <<= 1)
                ml = fmaxf(ml, __shfl_xor(ml, off, 16));
            float nm = fmaxf(rmax[v], ml);
            float alpha = __expf(rmax[v] - nm);
            rmax[v] = nm;
            float p0 = __expf(s[0][v] - nm);
            float p1 = __expf(s[1][v] - nm);
            float ps = p0 + p1;
#pragma unroll
            for (int off = 1; off < 16; off <<= 1)
                ps += __shfl_xor(ps, off, 16);
            rsum[v] = rsum[v] * alpha + ps;
#pragma unroll
            for (int c = 0; c < 4; ++c) occ[c][v] *= alpha;
            s[0][v] = p0; s[1][v] = p1;
        }

        // --- P round-trip through LDS into A-fragment layout ---
#pragma unroll
        for (int t = 0; t < 2; ++t)
#pragma unroll
            for (int v = 0; v < 8; ++v)
                Pb[wave][half * 8 + v][t * 16 + l16] = (bf16_t)s[t][v];
        // same-wave DS ops are in-order; no block barrier needed for Pb
        v16bf pfrag;
        {
            int kb = half * 8;
            v8bf lo = *(const v8bf*)(&Pb[wave][l16][kb]);
            v8bf hi = *(const v8bf*)(&Pb[wave][l16][kb + 16]);
            pfrag = cat16(lo, hi);
        }

        // --- O += P . V  (4 wmma over HD chunks of 16) ---
#pragma unroll
        for (int c = 0; c < 4; ++c) {
            int col = c * 16 + l16;               // HD column
            int kb2 = half * 8;                   // key offset
            v8bf lo = *(const v8bf*)(&Vt[col][kb2]);
            v8bf hi = *(const v8bf*)(&Vt[col][kb2 + 16]);
            occ[c] = __builtin_amdgcn_wmma_f32_16x16x32_bf16(
                false, pfrag, false, cat16(lo, hi), (short)0, occ[c],
                false, false);
        }
        __syncthreads();
    }

    // --- normalize and write to [M, A] bf16 (token-major for proj GEMM) ---
#pragma unroll
    for (int v = 0; v < 8; ++v) {
        float inv = 1.0f / rsum[v];
        int mrow = b * CN + qr0 + half * 8 + v;
#pragma unroll
        for (int c = 0; c < 4; ++c) {
            int col = h * CHD + c * 16 + l16;
            O[(size_t)mrow * CA + col] = (bf16_t)(occ[c][v] * inv);
        }
    }
}

// ---------------------------------------------------------------------------
// SwiGLU: hid[m,j] = silu(x12[m,j]) * x12[m, F+j], K-padded to Fp with zeros
// ---------------------------------------------------------------------------
__global__ void k_swiglu(const bf16_t* __restrict__ x12, bf16_t* __restrict__ hid) {
    int i = blockIdx.x * blockDim.x + threadIdx.x;
    if (i >= CM * CFP) return;
    int m = i / CFP, j = i % CFP;
    if (j < CF) {
        float a = (float)x12[(size_t)m * C2F + j];
        float g = (float)x12[(size_t)m * C2F + CF + j];
        hid[i] = (bf16_t)((a / (1.0f + __expf(-a))) * g);
    } else {
        hid[i] = (bf16_t)0.0f;
    }
}

// ---------------------------------------------------------------------------
// Host-side orchestration
// ---------------------------------------------------------------------------
extern "C" void kernel_launch(void* const* d_in, const int* in_sizes, int n_in,
                              void* d_out, int out_size, void* d_ws, size_t ws_size,
                              hipStream_t stream) {
    const float* x          = (const float*)d_in[0];
    const float* c          = (const float*)d_in[1];
    const float* norm1_w    = (const float*)d_in[2];
    const float* compress_w = (const float*)d_in[3];
    const float* compress_b = (const float*)d_in[4];
    const float* qkv_w      = (const float*)d_in[5];
    const float* qkv_b      = (const float*)d_in[6];
    const float* qn_w       = (const float*)d_in[7];
    const float* kn_w       = (const float*)d_in[8];
    const float* proj_w     = (const float*)d_in[9];
    const float* proj_b     = (const float*)d_in[10];
    const float* expand_w   = (const float*)d_in[11];
    const float* expand_b   = (const float*)d_in[12];
    const float* norm2_w    = (const float*)d_in[13];
    const float* w12_w      = (const float*)d_in[14];
    const float* w12_b      = (const float*)d_in[15];
    const float* w3_w       = (const float*)d_in[16];
    const float* w3_b       = (const float*)d_in[17];
    const float* ada_w      = (const float*)d_in[18];
    const float* ada_b      = (const float*)d_in[19];
    float* out = (float*)d_out;

    char* ws = (char*)d_ws;
    size_t off = 0;
    auto alloc = [&](size_t bytes) -> void* {
        void* p = ws + off;
        off = (off + bytes + 255) & ~(size_t)255;
        return p;
    };

    float*  silu_c = (float*)alloc((size_t)CB * CH * 4);
    float*  ada    = (float*)alloc((size_t)CB * SIXH * 4);
    bf16_t* cw_bf  = (bf16_t*)alloc((size_t)CA * CH * 2);
    bf16_t* qkvw_bf= (bf16_t*)alloc((size_t)3 * CA * CA * 2);
    bf16_t* pw_bf  = (bf16_t*)alloc((size_t)CA * CA * 2);
    bf16_t* ew_bf  = (bf16_t*)alloc((size_t)CH * CA * 2);
    bf16_t* w12_bf = (bf16_t*)alloc((size_t)C2F * CH * 2);
    bf16_t* w3_bf  = (bf16_t*)alloc((size_t)CH * CFP * 2);
    bf16_t* h_bf   = (bf16_t*)alloc((size_t)CM * CH * 2);      // also h2
    bf16_t* hc_bf  = (bf16_t*)alloc((size_t)CM * CA * 2);
    float*  qkvf   = (float*)alloc((size_t)CM * 3 * CA * 4);   // also x12 (bf16)
    bf16_t* qb     = (bf16_t*)alloc((size_t)3 * CM * CHD * CNH * 2); // q|k|v; also hid
    bf16_t* kb     = qb + (size_t)CM * CA;   // [B*NH, N, HD] flat = M*A elems each
    bf16_t* vb     = kb + (size_t)CM * CA;
    bf16_t* o_bf   = (bf16_t*)alloc((size_t)CM * CA * 2);
    bf16_t* p_bf   = (bf16_t*)alloc((size_t)CM * CA * 2);
    float*  xmid   = (float*)alloc((size_t)CM * CH * 4);
    bf16_t* x12_bf = (bf16_t*)qkvf;          // alias (qkvf dead after prep)
    bf16_t* hid_bf = qb;                     // alias (q/k/v dead after attn)
    (void)ws_size; (void)n_in; (void)in_sizes; (void)out_size;

    // 1) adaLN
    k_silu<<<(CB * CH) / 256, 256, 0, stream>>>(c, silu_c, CB * CH);
    k_ada<<<(CB * SIXH) / 256, 256, 0, stream>>>(silu_c, ada_w, ada_b, ada);

    // 2) weight conversions fp32 -> bf16
    k_cvt<<<(CA * CH) / 256, 256, 0, stream>>>(compress_w, cw_bf, CA * CH);
    k_cvt<<<(3 * CA * CA) / 256, 256, 0, stream>>>(qkv_w, qkvw_bf, 3 * CA * CA);
    k_cvt<<<(CA * CA) / 256, 256, 0, stream>>>(proj_w, pw_bf, CA * CA);
    k_cvt<<<(CH * CA) / 256, 256, 0, stream>>>(expand_w, ew_bf, CH * CA);
    k_cvt<<<(C2F * CH) / 256, 256, 0, stream>>>(w12_w, w12_bf, C2F * CH);
    k_cvt_pad<<<(CH * CFP) / 256, 256, 0, stream>>>(w3_w, w3_bf, CH, CF, CFP);

    // 3) attention branch
    k_modulate<<<CM, 256, 0, stream>>>(x, norm1_w, ada, 0, CH, h_bf);
    k_gemm<<<dim3(CA / GBN, CM / GBM), 256, 0, stream>>>(
        h_bf, cw_bf, compress_b, CM, CA, CH, nullptr, hc_bf,
        nullptr, nullptr, 0, 0);
    k_gemm<<<dim3((3 * CA) / GBN, CM / GBM), 256, 0, stream>>>(
        hc_bf, qkvw_bf, qkv_b, CM, 3 * CA, CA, qkvf, nullptr,
        nullptr, nullptr, 0, 1);
    k_qkvprep<<<CM * CNH, 32, 0, stream>>>(qkvf, qn_w, kn_w, qb, kb, vb, 0.125f);
    k_attn<<<dim3(CB * CNH, CN / 128), 256, 0, stream>>>(qb, kb, vb, o_bf);
    k_gemm<<<dim3(CA / GBN, CM / GBM), 256, 0, stream>>>(
        o_bf, pw_bf, proj_b, CM, CA, CA, nullptr, p_bf,
        nullptr, nullptr, 0, 0);
    k_gemm<<<dim3(CH / GBN, CM / GBM), 256, 0, stream>>>(
        p_bf, ew_bf, expand_b, CM, CH, CA, xmid, nullptr,
        x, ada + 2 * CH, SIXH, 2);

    // 4) SwiGLU FFN branch
    k_modulate<<<CM, 256, 0, stream>>>(xmid, norm2_w, ada, 3 * CH, 4 * CH, h_bf);
    k_gemm<<<dim3((C2F + GBN - 1) / GBN, CM / GBM), 256, 0, stream>>>(
        h_bf, w12_bf, w12_b, CM, C2F, CH, nullptr, x12_bf,
        nullptr, nullptr, 0, 0);
    k_swiglu<<<(CM * CFP) / 256, 256, 0, stream>>>(x12_bf, hid_bf);
    k_gemm<<<dim3(CH / GBN, CM / GBM), 256, 0, stream>>>(
        hid_bf, w3_bf, w3_b, CM, CH, CFP, out, nullptr,
        xmid, ada + 5 * CH, SIXH, 2);
}
